// TwoLayerGCN_14937896255857
// MI455X (gfx1250) — compile-verified, matrix operand order checked by reading, not run on hardware
//
#include <hip/hip_runtime.h>
#include <hip/hip_bf16.h>

#define NB 4
#define NN 2048
#define F1 128
#define H1 256
#define C2 64

typedef __attribute__((ext_vector_type(16))) __bf16 v16bf;
typedef __attribute__((ext_vector_type(8)))  float  v8f;

union Frag {
    v16bf v;
    uint4 u[2];
};

__device__ __forceinline__ unsigned short f2bf(float x) {
    unsigned u = __float_as_uint(x);
    unsigned r = u + 0x7FFFu + ((u >> 16) & 1u);   // round-to-nearest-even
    return (unsigned short)(r >> 16);
}

// 32-bit LDS byte offset of a generic pointer to __shared__ (addr[31:0] per ISA aperture rules)
__device__ __forceinline__ unsigned lds_off(const void* p) {
    return (unsigned)(unsigned long long)p;
}

// Async 16B copy global -> LDS, tracked by ASYNCcnt (gfx1250 VGLOBAL async op).
__device__ __forceinline__ void async_b128(const void* gaddr, unsigned ldsoff) {
    asm volatile("global_load_async_to_lds_b128 %0, %1, off"
                 :: "v"(ldsoff), "v"(gaddr)
                 : "memory");
}
__device__ __forceinline__ void wait_async0() {
    asm volatile("s_wait_asynccnt 0" ::: "memory");
}

// ---------------------------------------------------------------------------
// dinv[b*N+n] = rsqrt(1 + sum_j A[b,n,j])   (one wave32 per row)
// ---------------------------------------------------------------------------
__global__ __launch_bounds__(256) void degree_kernel(const float* __restrict__ A,
                                                     float* __restrict__ dinv) {
    const int row  = (blockIdx.x * blockDim.x + threadIdx.x) >> 5; // global row
    const int lane = threadIdx.x & 31;
    const float* __restrict__ Arow = A + (size_t)row * NN;
    float s = 0.f;
    for (int j = lane; j < NN; j += 32) s += Arow[j];
    for (int o = 16; o; o >>= 1) s += __shfl_down(s, o, 32);
    if (lane == 0) dinv[row] = rsqrtf(s + 1.0f);
}

// ---------------------------------------------------------------------------
// fs[n] = h[n,:]·asrc, fd[n] = h[n,:]·adst   (one wave32 per node)
// ---------------------------------------------------------------------------
__global__ __launch_bounds__(256) void proj_kernel(const float* __restrict__ h,
                                                   const float* __restrict__ asrc,
                                                   const float* __restrict__ adst,
                                                   float* __restrict__ fs,
                                                   float* __restrict__ fd, int K) {
    const int node = (blockIdx.x * blockDim.x + threadIdx.x) >> 5;
    const int lane = threadIdx.x & 31;
    const float* __restrict__ row = h + (size_t)node * K;
    float s = 0.f, d = 0.f;
    for (int k = lane; k < K; k += 32) {
        float v = row[k];
        s += v * asrc[k];
        d += v * adst[k];
    }
    for (int o = 16; o; o >>= 1) { s += __shfl_down(s, o, 32); d += __shfl_down(d, o, 32); }
    if (lane == 0) { fs[node] = s; fd[node] = d; }
}

__global__ void zero_kernel(float* __restrict__ p, int n) {
    int i = blockIdx.x * blockDim.x + threadIdx.x;
    if (i < n) p[i] = 0.f;
}

// ---------------------------------------------------------------------------
// Fused masked-softmax attention * normalized adjacency, times features:
//   Z[b, i, :] = sum_j ( att[i,j] * dinv_i * (A_ij + delta_ij) * dinv_j ) * H[b, j, :]
// One block = 16 rows of one batch. K-loop over j in chunks of 32.
// Raw f32 tiles are pulled with GLOBAL_LOAD_ASYNC_TO_LDS_B128 (ASYNCcnt),
// converted branchlessly to bf16 LDS tiles, then consumed by
// v_wmma_f32_16x16x32_bf16. Wave w produces feature columns [32w, 32w+32).
// ---------------------------------------------------------------------------
template <int FDIM, int T>
__global__ __launch_bounds__(T) void fused_attn_spmm(const float* __restrict__ A,
                                                     const float* __restrict__ Hsrc,
                                                     const float* __restrict__ fsrc,
                                                     const float* __restrict__ fdst,
                                                     const float* __restrict__ dinv,
                                                     float* __restrict__ Z) {
    constexpr int TPR = T / 16;       // threads per row in stats pass
    const int b    = blockIdx.x / (NN / 16);
    const int row0 = (blockIdx.x % (NN / 16)) * 16;
    const int tid  = threadIdx.x;
    const size_t bN = (size_t)b * NN;

    __shared__ float s_fsrc[16];
    __shared__ float s_scale[16];                           // (1/denom_i) * dinv_i
    __shared__ alignas(16) float sAraw[16 * 32];            // raw f32 A tile (async dest)
    __shared__ alignas(16) float sXraw[32 * FDIM];          // raw f32 feature tile (async dest)
    __shared__ alignas(16) unsigned short sM[16 * 40];      // bf16 M tile, 80B row stride
    __shared__ alignas(16) unsigned short sX[FDIM * 40];    // bf16 feature tile (transposed)

    if (tid < 16) {
        s_fsrc[tid]  = fsrc[bN + row0 + tid];
        s_scale[tid] = 0.f;
    }
    __syncthreads();

    // ---- pass 1: softmax denominators (exp without max-shift; inputs are O(1)) ----
    {
        const int r  = tid / TPR;
        const int jl = tid % TPR;
        const float fsr = s_fsrc[r];
        const float* __restrict__ Arow = A + (bN + row0 + r) * NN;
        float acc = 0.f;
        for (int j = jl; j < NN; j += TPR) {
            __builtin_prefetch(Arow + j + TPR * 16, 0, 1);
            float a = Arow[j];
            float e = fsr + fdst[bN + j];
            e = e > 0.f ? e : 0.2f * e;               // leaky_relu(0.2)
            acc += (a > 0.f) ? __expf(e) : 0.f;       // branchless mask
        }
        atomicAdd(&s_scale[r], acc);                  // ds_add_f32
    }
    __syncthreads();
    if (tid < 16) {
        float dnm = s_scale[tid];
        s_scale[tid] = (dnm > 0.f ? 1.0f / dnm : 0.f) * dinv[bN + row0 + tid];
    }
    __syncthreads();

    // ---- pass 2: async-staged tiles -> bf16 -> WMMA ----
    const int lane = tid & 31;
    const int warp = tid >> 5;
    const int koff = (lane & 16) ? 8 : 0;             // per-lane K offset (A/B frag layout)
    const int mrow = lane & 15;
    const unsigned sAoff = lds_off(sAraw);
    const unsigned sXoff = lds_off(sXraw);

    v8f acc0 = {}; v8f acc1 = {};

    for (int kc = 0; kc < NN; kc += 32) {
        // ---- async copy raw tiles into LDS (16B per lane per issue) ----
        {
            const float* __restrict__ Abase = A + (bN + row0) * NN + kc;  // 16 rows x 128B
            for (int e = tid; e < 16 * 8; e += T) {
                const int r = e >> 3, c = e & 7;
                async_b128(Abase + (size_t)r * NN + c * 4, sAoff + r * 128 + c * 16);
            }
            // 32 feature rows are contiguous in memory: one linear 32*FDIM*4B region
            const float* __restrict__ Xbase = Hsrc + (bN + kc) * FDIM;
            for (int e = tid; e < 8 * FDIM; e += T) {
                async_b128(Xbase + e * 4, sXoff + e * 16);
            }
            wait_async0();
        }
        __syncthreads();   // raw tiles visible to all waves

        // ---- convert: M tile [16 x 32] -> bf16 (branchless) ----
        for (int e = tid; e < 16 * 32; e += T) {
            const int r = e >> 5, k = e & 31;
            const int j = kc + k;
            const float a = sAraw[e];
            float ee = s_fsrc[r] + fdst[bN + j];
            ee = ee > 0.f ? ee : 0.2f * ee;
            const float ai = a + ((row0 + r) == j ? 1.f : 0.f);
            const float w = __expf(ee) * s_scale[r] * dinv[bN + j] * ai;
            sM[r * 40 + k] = f2bf(a > 0.f ? w : 0.f);
        }
        // ---- convert: feature tile transposed sX[n][k] = H[b, kc+k, n] ----
        for (int e = tid; e < FDIM * 32; e += T) {
            const int k = e / FDIM, n = e % FDIM;
            sX[n * 40 + k] = f2bf(sXraw[k * FDIM + n]);
        }
        __syncthreads();   // bf16 tiles ready; raw buffers free for next iteration

        Frag fa;
        fa.u[0] = *(const uint4*)&sM[mrow * 40 + koff];
        fa.u[1] = *(const uint4*)&sM[mrow * 40 + 16 + koff];

        const int n0 = warp * 32 + mrow;
        Frag fb0, fb1;
        fb0.u[0] = *(const uint4*)&sX[n0 * 40 + koff];
        fb0.u[1] = *(const uint4*)&sX[n0 * 40 + 16 + koff];
        fb1.u[0] = *(const uint4*)&sX[(n0 + 16) * 40 + koff];
        fb1.u[1] = *(const uint4*)&sX[(n0 + 16) * 40 + 16 + koff];

        acc0 = __builtin_amdgcn_wmma_f32_16x16x32_bf16(false, fa.v, false, fb0.v,
                                                       (short)0, acc0, false, false);
        acc1 = __builtin_amdgcn_wmma_f32_16x16x32_bf16(false, fa.v, false, fb1.v,
                                                       (short)0, acc1, false, false);
    }

    // C/D layout: VGPR r -> row = r + 8*(lane>=16), col = lane&15
    const int rbase = (lane >> 4) << 3;
    const int c0 = warp * 32 + (lane & 15);
    for (int r = 0; r < 8; ++r) {
        const size_t gr = bN + row0 + rbase + r;
        Z[gr * FDIM + c0]      = acc0[r];
        Z[gr * FDIM + c0 + 16] = acc1[r];
    }
}

// ---------------------------------------------------------------------------
// out[row, :] = act(Zin[row, :KD] @ W[KD, NO] + bias);  optional graph-sum.
// One block = 16 rows; NO/32 waves; bf16 WMMA with f32 accumulate.
// ---------------------------------------------------------------------------
template <int KD, int NO, bool RELU, bool GSUM, int T>
__global__ __launch_bounds__(T) void gemm_bias_act(const float* __restrict__ Zin,
                                                   const float* __restrict__ W,
                                                   const float* __restrict__ bias,
                                                   float* __restrict__ out,
                                                   float* __restrict__ gout) {
    const int row0 = blockIdx.x * 16;
    const int tid  = threadIdx.x;
    const int lane = tid & 31;
    const int warp = tid >> 5;
    const int koff = (lane & 16) ? 8 : 0;
    const int mrow = lane & 15;

    __shared__ alignas(16) unsigned short sZ[16 * 40];
    __shared__ alignas(16) unsigned short sW[NO * 40];
    __shared__ float s_gs[NO];

    if (GSUM) {
        for (int i = tid; i < NO; i += T) s_gs[i] = 0.f;
    }

    v8f acc0 = {}; v8f acc1 = {};
    for (int kc = 0; kc < KD; kc += 32) {
        for (int e = tid; e < 16 * 32; e += T) {
            const int r = e >> 5, k = e & 31;
            sZ[r * 40 + k] = f2bf(Zin[(size_t)(row0 + r) * KD + kc + k]);
        }
        for (int e = tid; e < NO * 32; e += T) {
            const int k = e / NO, n = e % NO;
            sW[n * 40 + k] = f2bf(W[(size_t)(kc + k) * NO + n]);
        }
        __syncthreads();

        Frag fa;
        fa.u[0] = *(const uint4*)&sZ[mrow * 40 + koff];
        fa.u[1] = *(const uint4*)&sZ[mrow * 40 + 16 + koff];

        const int n0 = warp * 32 + mrow;
        Frag fb0, fb1;
        fb0.u[0] = *(const uint4*)&sW[n0 * 40 + koff];
        fb0.u[1] = *(const uint4*)&sW[n0 * 40 + 16 + koff];
        fb1.u[0] = *(const uint4*)&sW[(n0 + 16) * 40 + koff];
        fb1.u[1] = *(const uint4*)&sW[(n0 + 16) * 40 + 16 + koff];

        acc0 = __builtin_amdgcn_wmma_f32_16x16x32_bf16(false, fa.v, false, fb0.v,
                                                       (short)0, acc0, false, false);
        acc1 = __builtin_amdgcn_wmma_f32_16x16x32_bf16(false, fa.v, false, fb1.v,
                                                       (short)0, acc1, false, false);
        __syncthreads();
    }

    const int rbase = (lane >> 4) << 3;
    const int c0 = warp * 32 + (lane & 15);
    const float bi0 = bias[c0], bi1 = bias[c0 + 16];
    for (int r = 0; r < 8; ++r) {
        const int gr = row0 + rbase + r;
        float v0 = acc0[r] + bi0;
        float v1 = acc1[r] + bi1;
        if (RELU) { v0 = fmaxf(v0, 0.f); v1 = fmaxf(v1, 0.f); }
        out[(size_t)gr * NO + c0]      = v0;
        out[(size_t)gr * NO + c0 + 16] = v1;
        if (GSUM) { atomicAdd(&s_gs[c0], v0); atomicAdd(&s_gs[c0 + 16], v1); }
    }
    if (GSUM) {
        __syncthreads();
        const int bb = row0 / NN;
        for (int i = tid; i < NO; i += T) atomicAdd(&gout[bb * NO + i], s_gs[i]);
    }
}

// ---------------------------------------------------------------------------
extern "C" void kernel_launch(void* const* d_in, const int* in_sizes, int n_in,
                              void* d_out, int out_size, void* d_ws, size_t ws_size,
                              hipStream_t stream) {
    const float* X   = (const float*)d_in[0];
    const float* A   = (const float*)d_in[1];
    const float* a1s = (const float*)d_in[2];
    const float* a1d = (const float*)d_in[3];
    const float* W1  = (const float*)d_in[4];
    const float* b1  = (const float*)d_in[5];
    const float* a2s = (const float*)d_in[6];
    const float* a2d = (const float*)d_in[7];
    const float* W2  = (const float*)d_in[8];
    const float* b2  = (const float*)d_in[9];

    float* nodes = (float*)d_out;                       // [B, N, C]
    float* gout  = nodes + (size_t)NB * NN * C2;        // [B, C]

    float* ws   = (float*)d_ws;
    float* dinv = ws;  ws += (size_t)NB * NN;
    float* f1s  = ws;  ws += (size_t)NB * NN;
    float* f1d  = ws;  ws += (size_t)NB * NN;
    float* f2s  = ws;  ws += (size_t)NB * NN;
    float* f2d  = ws;  ws += (size_t)NB * NN;
    float* Z1   = ws;  ws += (size_t)NB * NN * F1;      // M1 @ X
    float* hid  = ws;  ws += (size_t)NB * NN * H1;      // layer-1 output
    float* Z2   = ws;  ws += (size_t)NB * NN * H1;      // M2 @ hid

    zero_kernel<<<1, 256, 0, stream>>>(gout, NB * C2);

    degree_kernel<<<(NB * NN) / 8, 256, 0, stream>>>(A, dinv);
    proj_kernel<<<(NB * NN) / 8, 256, 0, stream>>>(X, a1s, a1d, f1s, f1d, F1);

    fused_attn_spmm<F1, 128><<<NB * (NN / 16), 128, 0, stream>>>(A, X, f1s, f1d, dinv, Z1);
    gemm_bias_act<F1, H1, true, false, 256><<<(NB * NN) / 16, 256, 0, stream>>>(Z1, W1, b1, hid, nullptr);

    proj_kernel<<<(NB * NN) / 8, 256, 0, stream>>>(hid, a2s, a2d, f2s, f2d, H1);

    fused_attn_spmm<H1, 256><<<NB * (NN / 16), 256, 0, stream>>>(A, hid, f2s, f2d, dinv, Z2);
    gemm_bias_act<H1, C2, false, true, 64><<<(NB * NN) / 16, 64, 0, stream>>>(Z2, W2, b2, nodes, gout);
}
